// SwinCrossBlock_25159918420369
// MI455X (gfx1250) — matrix operand order repaired and checked by reading, not compiled, not run on hardware
//
#include <hip/hip_runtime.h>
#include <hip/hip_bf16.h>
#include <math.h>

typedef __attribute__((ext_vector_type(16))) _Float16 v16h;
typedef __attribute__((ext_vector_type(8)))  _Float16 v8h;
typedef __attribute__((ext_vector_type(8)))  float    v8f;
typedef int v4i __attribute__((vector_size(16)));

#define BDIM 16
#define HDIM 56
#define WDIM 56
#define CDIM 256
#define NHEADS 8
#define WS 7
#define NWIN 49                     // tokens per window
#define HEAD_DIM 32
#define HIDN 1024
#define NTOK (BDIM*HDIM*WDIM)       // 50176 tokens per tensor
#define NWINDOWS (BDIM*64)          // 1024 windows total
#define QK_SCALE 0.17677669529663687f

enum { EPI_QKV = 0, EPI_PROJ = 1, EPI_FC1 = 2, EPI_FC2 = 3 };

#if defined(__AMDGCN__) && __has_builtin(__builtin_amdgcn_global_load_async_to_lds_b128)
#define ASYNC_LDS 1
#else
#define ASYNC_LDS 0
#endif

// 16-byte async global->LDS copy (gfx1250 GLOBAL_LOAD_ASYNC_TO_LDS_B128)
static __device__ __forceinline__ void async_cp16(const _Float16* g, _Float16* l) {
#if ASYNC_LDS
  __builtin_amdgcn_global_load_async_to_lds_b128(
      (__attribute__((address_space(1))) v4i*)(void*)g,
      (__attribute__((address_space(3))) v4i*)(void*)l, 0, 0);
#else
  (void)g; (void)l;
#endif
}
static __device__ __forceinline__ void async_wait0() {
#if ASYNC_LDS
#if __has_builtin(__builtin_amdgcn_s_wait_asynccnt)
  __builtin_amdgcn_s_wait_asynccnt(0);
#else
  asm volatile("s_wait_asynccnt 0x0" ::: "memory");
#endif
#endif
}

// build a v16h WMMA fragment from two contiguous 16B LDS chunks
static __device__ __forceinline__ v16h frag16(const _Float16* p0, const _Float16* p1) {
  v8h a = *(const v8h*)p0;
  v8h b = *(const v8h*)p1;
  return __builtin_shufflevector(a, b, 0, 1, 2, 3, 4, 5, 6, 7,
                                 8, 9, 10, 11, 12, 13, 14, 15);
}

// source token (b,h,w) -> row in rolled(-3,-3) + window-partitioned order
static __device__ __forceinline__ int wrow_from_token(int tok) {
  int b  = tok / (HDIM * WDIM);
  int hw = tok % (HDIM * WDIM);
  int h = hw / WDIM, w = hw % WDIM;
  int hr = (h + HDIM - 3) % HDIM;
  int wr = (w + WDIM - 3) % WDIM;
  int wi = hr / WS, ir = hr % WS;
  int wj = wr / WS, jc = wr % WS;
  return ((b * 64 + wi * 8 + wj) * NWIN) + ir * WS + jc;
}

// ---------- fp32 -> fp16 weight conversion WITH transpose: dst[N][K] ----------
__global__ void k_cvt_t(const float* __restrict__ src, _Float16* __restrict__ dst,
                        int K, int N) {
  int i = blockIdx.x * 256 + threadIdx.x;
  if (i < K * N) {
    int k = i / N, n = i % N;
    dst[(size_t)n * K + k] = (_Float16)src[i];
  }
}

// ---------------- LayerNorm (one token per 256-thread block) ----------------
__global__ __launch_bounds__(256) void k_ln(const float* __restrict__ x,
                                            const float* __restrict__ gam,
                                            const float* __restrict__ bet,
                                            _Float16* __restrict__ out,
                                            int window_map) {
  int tok = blockIdx.x;
  int c = threadIdx.x;
  __shared__ float red[256];
  float v = x[(size_t)tok * CDIM + c];
  red[c] = v;
  __syncthreads();
  for (int s = 128; s > 0; s >>= 1) { if (c < s) red[c] += red[c + s]; __syncthreads(); }
  float mu = red[0] * (1.0f / CDIM);
  __syncthreads();
  float d = v - mu;
  red[c] = d * d;
  __syncthreads();
  for (int s = 128; s > 0; s >>= 1) { if (c < s) red[c] += red[c + s]; __syncthreads(); }
  float var = red[0] * (1.0f / CDIM);
  float y = d * rsqrtf(var + 1e-5f) * gam[c] + bet[c];
  int row = window_map ? wrow_from_token(tok) : tok;
  out[(size_t)row * CDIM + c] = (_Float16)y;
}

// ---------------- Tiled WMMA GEMM with fused epilogues ----------------
// C[M,N] = A[M,K](f16) * B[K,N] + bias, with B supplied TRANSPOSED: BwT[N][K].
// Block tile 64x64, 4 waves of 32x32 (2x2 WMMA tiles each), BK=32,
// double-buffered LDS with one barrier per k-step and async global->LDS copies.
__global__ __launch_bounds__(128) void k_gemm(
    const _Float16* __restrict__ A, const _Float16* __restrict__ BwT,
    const float* __restrict__ bias,
    int M, int K, int N, int epi, int dir,
    _Float16* __restrict__ out_h, float* __restrict__ out_f,
    const float* __restrict__ shortcut) {
  __shared__ __align__(16) _Float16 As[2][64][32];   // [buf][m][k]
  __shared__ __align__(16) _Float16 Bs[2][64][32];   // [buf][n][k]
  const int tid = threadIdx.x;
  const int lane = tid & 31, wid = tid >> 5;
  const int wm = wid >> 1, wn = wid & 1;
  const int m0 = blockIdx.x * 64, n0 = blockIdx.y * 64;
  const int l15 = lane & 15, lh = lane >> 4;

  v8f acc[2][2] = {};

  const int lr = tid >> 2;               // 0..31 -> rows lr and lr+32
  const int lc = (tid & 3) * 8;          // 16B chunk within a 32-half row

  const _Float16* gA0 = &A[(size_t)(m0 + lr) * K + lc];
  const _Float16* gA1 = &A[(size_t)(m0 + lr + 32) * K + lc];
  const _Float16* gB0 = &BwT[(size_t)(n0 + lr) * K + lc];
  const _Float16* gB1 = &BwT[(size_t)(n0 + lr + 32) * K + lc];

#if ASYNC_LDS
  async_cp16(gA0, &As[0][lr][lc]);
  async_cp16(gA1, &As[0][lr + 32][lc]);
  async_cp16(gB0, &Bs[0][lr][lc]);
  async_cp16(gB1, &Bs[0][lr + 32][lc]);
#else
  v8h ra0 = *(const v8h*)gA0;
  v8h ra1 = *(const v8h*)gA1;
  v8h rb0 = *(const v8h*)gB0;
  v8h rb1 = *(const v8h*)gB1;
#endif

  for (int k0 = 0; k0 < K; k0 += 32) {
    const int buf = (k0 >> 5) & 1;
#if ASYNC_LDS
    async_wait0();                       // my async copies into `buf` done
    __syncthreads();                     // everyone's copies visible
    if (k0 + 32 < K) {                   // fill other buffer while we compute
      async_cp16(gA0 + k0 + 32, &As[buf ^ 1][lr][lc]);
      async_cp16(gA1 + k0 + 32, &As[buf ^ 1][lr + 32][lc]);
      async_cp16(gB0 + k0 + 32, &Bs[buf ^ 1][lr][lc]);
      async_cp16(gB1 + k0 + 32, &Bs[buf ^ 1][lr + 32][lc]);
      if (k0 + 64 < K) {
        __builtin_prefetch(gA0 + k0 + 64, 0, 1);
        __builtin_prefetch(gB0 + k0 + 64, 0, 1);
      }
    }
#else
    *(v8h*)&As[buf][lr][lc]      = ra0;  // registers -> LDS (loads were batched)
    *(v8h*)&As[buf][lr + 32][lc] = ra1;
    *(v8h*)&Bs[buf][lr][lc]      = rb0;
    *(v8h*)&Bs[buf][lr + 32][lc] = rb1;
    __syncthreads();
    if (k0 + 32 < K) {                   // batch next loads, overlap with WMMAs
      ra0 = *(const v8h*)(gA0 + k0 + 32);
      ra1 = *(const v8h*)(gA1 + k0 + 32);
      rb0 = *(const v8h*)(gB0 + k0 + 32);
      rb1 = *(const v8h*)(gB1 + k0 + 32);
      if (k0 + 64 < K) {
        __builtin_prefetch(gA0 + k0 + 64, 0, 1);
        __builtin_prefetch(gB0 + k0 + 64, 0, 1);
      }
    }
#endif

    v16h af[2];
    #pragma unroll
    for (int mt = 0; mt < 2; ++mt) {
      const _Float16* pr = &As[buf][wm * 32 + mt * 16 + l15][0];
      af[mt] = frag16(pr + lh * 8, pr + 16 + lh * 8);
    }
    #pragma unroll
    for (int nt = 0; nt < 2; ++nt) {
      const _Float16* pc = &Bs[buf][wn * 32 + nt * 16 + l15][0];
      v16h bf = frag16(pc + lh * 8, pc + 16 + lh * 8);
      #pragma unroll
      for (int mt = 0; mt < 2; ++mt)
        acc[mt][nt] = __builtin_amdgcn_wmma_f32_16x16x32_f16(
            false, af[mt], false, bf, (short)0, acc[mt][nt], false, false);
    }
  }

  #pragma unroll
  for (int mt = 0; mt < 2; ++mt)
  #pragma unroll
  for (int nt = 0; nt < 2; ++nt)
  #pragma unroll
  for (int r = 0; r < 8; ++r) {
    int row = m0 + wm * 32 + mt * 16 + r + lh * 8;
    int col = n0 + wn * 32 + nt * 16 + l15;
    if (row >= M) continue;
    float v = acc[mt][nt][r] + bias[col];
    if (epi == EPI_QKV) {
      int t = col >> 8;                 // 0=q 1=k 2=v
      int head = (col & 255) >> 5;
      int dd = col & 31;
      if (t == 0) v *= QK_SCALE;
      int win = row / NWIN, i = row % NWIN;
      size_t o = ((((size_t)(dir * 3 + t) * NWINDOWS + win) * NHEADS + head) * NWIN + i) * HEAD_DIM + dd;
      out_h[o] = (_Float16)v;
    } else if (epi == EPI_PROJ) {
      int win = row / NWIN, i = row % NWIN;
      int b = win >> 6, wl = win & 63;
      int wi = wl >> 3, wj = wl & 7;
      int hh = (wi * WS + i / WS + 3) % HDIM;   // roll back (+3,+3)
      int ww = (wj * WS + i % WS + 3) % WDIM;
      size_t tok = (size_t)b * (HDIM * WDIM) + hh * WDIM + ww;
      out_f[tok * CDIM + col] = v + shortcut[tok * CDIM + col];
    } else if (epi == EPI_FC1) {
      float u = v + 0.044715f * v * v * v;
      float gl = 0.5f * v * (1.0f + tanhf(0.7978845608028654f * u));
      out_h[(size_t)row * HIDN + col] = (_Float16)gl;
    } else {                                    // EPI_FC2: residual accumulate
      out_f[(size_t)row * CDIM + col] += v;
    }
  }
}

// ---------------- Windowed cross-attention, one (window,head,dir) per block ----
__global__ __launch_bounds__(128) void k_attn(const _Float16* __restrict__ qkv,
                                              const float* __restrict__ rel_table,
                                              _Float16* __restrict__ attn_out) {
  const int head = blockIdx.x & 7;
  const int win  = blockIdx.x >> 3;
  const int dir  = blockIdx.y;                  // 0: q<-x1,kv<-x2 ; 1: swapped
  __shared__ __align__(16) _Float16 Qs[64][32];   // [i][d]
  __shared__ __align__(16) _Float16 Ks[64][32];   // [j][d] == transposed B layout
  __shared__ __align__(16) _Float16 Vt[32][64];   // [d][j] transposed V
  __shared__ __align__(16) _Float16 Ps[64][72];   // fp16 probabilities
  __shared__ float S[64][65];
  const int tid = threadIdx.x, lane = tid & 31, wid = tid >> 5;
  const int l15 = lane & 15, lh = lane >> 4;

  const size_t bq = (((size_t)(dir * 3 + 0) * NWINDOWS + win) * NHEADS + head) * (size_t)NWIN * HEAD_DIM;
  const size_t bk = (((size_t)((dir ^ 1) * 3 + 1) * NWINDOWS + win) * NHEADS + head) * (size_t)NWIN * HEAD_DIM;
  const size_t bv = (((size_t)((dir ^ 1) * 3 + 2) * NWINDOWS + win) * NHEADS + head) * (size_t)NWIN * HEAD_DIM;

  #pragma unroll
  for (int it = 0; it < 2; ++it) {
    int idx = tid + it * 128;
    int r = idx >> 2, c = (idx & 3) * 8;
    v8h z = {};
#if ASYNC_LDS
    if (r < NWIN) {
      async_cp16(&qkv[bq + (size_t)r * 32 + c], &Qs[r][c]);
      async_cp16(&qkv[bk + (size_t)r * 32 + c], &Ks[r][c]);
    } else {
      *(v8h*)&Qs[r][c] = z;
      *(v8h*)&Ks[r][c] = z;
    }
#else
    *(v8h*)&Qs[r][c] = (r < NWIN) ? *(const v8h*)&qkv[bq + (size_t)r * 32 + c] : z;
    *(v8h*)&Ks[r][c] = (r < NWIN) ? *(const v8h*)&qkv[bk + (size_t)r * 32 + c] : z;
#endif
    v8h vv = (r < NWIN) ? *(const v8h*)&qkv[bv + (size_t)r * 32 + c] : z;
    #pragma unroll
    for (int u = 0; u < 8; ++u) Vt[c + u][r] = vv[u];   // transpose V into LDS
  }
  async_wait0();
  __syncthreads();

  // S = Q * K^T : each wave owns a 16-row tile, K-dim = 32 (one WMMA step)
  const _Float16* pq = &Qs[wid * 16 + l15][0];
  v16h aq = frag16(pq + lh * 8, pq + 16 + lh * 8);
  #pragma unroll
  for (int nt = 0; nt < 4; ++nt) {
    const _Float16* pk = &Ks[nt * 16 + l15][0];
    v16h bkf = frag16(pk + lh * 8, pk + 16 + lh * 8);
    v8f s = {};
    s = __builtin_amdgcn_wmma_f32_16x16x32_f16(false, aq, false, bkf, (short)0, s, false, false);
    #pragma unroll
    for (int r = 0; r < 8; ++r) S[wid * 16 + r + lh * 8][nt * 16 + l15] = s[r];
  }
  __syncthreads();

  // fused rel-pos bias + shift mask + softmax; two lanes per row (32 cols each)
  const int row = wid * 16 + l15;
  const int c0 = lh * 32;
  const int i = row;
  const int ri = i / WS, ci = i % WS;
  const int wl = win & 63, wib = wl >> 3, wjb = wl & 7;
  int regi = 0;
  if (i < NWIN) {
    int hh = wib * WS + ri, ww = wjb * WS + ci;
    int rh = hh < HDIM - WS ? 0 : (hh < HDIM - 3 ? 1 : 2);
    int rw = ww < WDIM - WS ? 0 : (ww < WDIM - 3 ? 1 : 2);
    regi = rh * 3 + rw;
  }
  float mx = -1e30f;
  for (int cc = 0; cc < 32; ++cc) {
    int j = c0 + cc;
    float s = S[row][j];
    if (j >= NWIN) s = -1e30f;
    else if (i < NWIN) {
      int rj = j / WS, cj = j % WS;
      s += rel_table[((ri - rj + WS - 1) * (2 * WS - 1) + (ci - cj + WS - 1)) * NHEADS + head];
      int hh = wib * WS + rj, ww = wjb * WS + cj;
      int rh = hh < HDIM - WS ? 0 : (hh < HDIM - 3 ? 1 : 2);
      int rw = ww < WDIM - WS ? 0 : (ww < WDIM - 3 ? 1 : 2);
      if (rh * 3 + rw != regi) s -= 100.0f;
    }
    S[row][j] = s;
    mx = fmaxf(mx, s);
  }
  mx = fmaxf(mx, __shfl_xor(mx, 16));
  float sum = 0.0f;
  for (int cc = 0; cc < 32; ++cc) {
    int j = c0 + cc;
    float e = __expf(S[row][j] - mx);
    S[row][j] = e;
    sum += e;
  }
  sum += __shfl_xor(sum, 16);
  float inv = 1.0f / sum;
  for (int cc = 0; cc < 32; ++cc) {
    int j = c0 + cc;
    Ps[row][j] = (_Float16)(S[row][j] * inv);   // fp16 P for the next WMMA
  }

  // O = P * V : K-dim 64 (2 steps), N = 32 (2 tiles)
  v8f o[2] = {};
  #pragma unroll
  for (int ks = 0; ks < 2; ++ks) {
    const _Float16* pp = &Ps[wid * 16 + l15][ks * 32];
    v16h ap = frag16(pp + lh * 8, pp + 16 + lh * 8);
    #pragma unroll
    for (int nt = 0; nt < 2; ++nt) {
      const _Float16* pv = &Vt[nt * 16 + l15][ks * 32];
      v16h bvf = frag16(pv + lh * 8, pv + 16 + lh * 8);
      o[nt] = __builtin_amdgcn_wmma_f32_16x16x32_f16(false, ap, false, bvf, (short)0, o[nt], false, false);
    }
  }
  #pragma unroll
  for (int nt = 0; nt < 2; ++nt)
  #pragma unroll
  for (int r = 0; r < 8; ++r) {
    int m = wid * 16 + r + lh * 8;
    if (m < NWIN)
      attn_out[((size_t)dir * NTOK + (size_t)win * NWIN + m) * CDIM +
               head * HEAD_DIM + nt * 16 + l15] = (_Float16)o[nt][r];
  }
}

extern "C" void kernel_launch(void* const* d_in, const int* in_sizes, int n_in,
                              void* d_out, int out_size, void* d_ws, size_t ws_size,
                              hipStream_t stream) {
  (void)in_sizes; (void)n_in; (void)out_size; (void)ws_size;
  const float* x1     = (const float*)d_in[0];
  const float* x2     = (const float*)d_in[1];
  const float* n1g    = (const float*)d_in[2];
  const float* n1b    = (const float*)d_in[3];
  const float* qkv_w  = (const float*)d_in[4];
  const float* qkv_b  = (const float*)d_in[5];
  const float* proj_w = (const float*)d_in[6];
  const float* proj_b = (const float*)d_in[7];
  const float* relt   = (const float*)d_in[8];
  const float* n2g    = (const float*)d_in[9];
  const float* n2b    = (const float*)d_in[10];
  const float* fc1_w  = (const float*)d_in[11];
  const float* fc1_b  = (const float*)d_in[12];
  const float* fc2_w  = (const float*)d_in[13];
  const float* fc2_b  = (const float*)d_in[14];
  float* out = (float*)d_out;

  char* ws = (char*)d_ws;
  size_t off = 0;
  _Float16* wbuf   = (_Float16*)(ws + off); off += (size_t)2 * NTOK * CDIM * 2;
  _Float16* qkvw_t = (_Float16*)(ws + off); off += (size_t)CDIM * 3 * CDIM * 2;
  _Float16* projw_t= (_Float16*)(ws + off); off += (size_t)CDIM * CDIM * 2;
  _Float16* fc1w_t = (_Float16*)(ws + off); off += (size_t)CDIM * HIDN * 2;
  _Float16* fc2w_t = (_Float16*)(ws + off); off += (size_t)HIDN * CDIM * 2;
  // union region: QKV (154MB) + attn_out (51MB) later reused as fc1 hidden (205.5MB)
  _Float16* qkvbuf  = (_Float16*)(ws + off);
  size_t qkv_bytes  = (size_t)2 * 3 * NWINDOWS * NHEADS * NWIN * HEAD_DIM * 2;
  _Float16* attnout = (_Float16*)(ws + off + qkv_bytes);
  _Float16* hidden  = (_Float16*)(ws + off);

  // 1. weights -> fp16, transposed to [N][K]
  k_cvt_t<<<(CDIM*3*CDIM + 255)/256, 256, 0, stream>>>(qkv_w,  qkvw_t,  CDIM, 3*CDIM);
  k_cvt_t<<<(CDIM*CDIM   + 255)/256, 256, 0, stream>>>(proj_w, projw_t, CDIM, CDIM);
  k_cvt_t<<<(CDIM*HIDN   + 255)/256, 256, 0, stream>>>(fc1_w,  fc1w_t,  CDIM, HIDN);
  k_cvt_t<<<(HIDN*CDIM   + 255)/256, 256, 0, stream>>>(fc2_w,  fc2w_t,  HIDN, CDIM);

  // 2. LN1 (fused roll + window partition into row order)
  k_ln<<<NTOK, 256, 0, stream>>>(x1, n1g, n1b, wbuf, 1);
  k_ln<<<NTOK, 256, 0, stream>>>(x2, n1g, n1b, wbuf + (size_t)NTOK * CDIM, 1);

  // 3. QKV projections (scatter to per-head layout, Q pre-scaled)
  dim3 gqkv(NTOK / 64, (3 * CDIM) / 64);
  k_gemm<<<gqkv, 128, 0, stream>>>(wbuf, qkvw_t, qkv_b, NTOK, CDIM, 3*CDIM,
                                   EPI_QKV, 0, qkvbuf, nullptr, nullptr);
  k_gemm<<<gqkv, 128, 0, stream>>>(wbuf + (size_t)NTOK * CDIM, qkvw_t, qkv_b,
                                   NTOK, CDIM, 3*CDIM, EPI_QKV, 1, qkvbuf, nullptr, nullptr);

  // 4. both cross-attention directions
  k_attn<<<dim3(NWINDOWS * NHEADS, 2), 128, 0, stream>>>(qkvbuf, relt, attnout);

  // 5. proj + inverse roll/window scatter + residual -> d_out holds x' (f32)
  dim3 gpro(NTOK / 64, CDIM / 64);
  k_gemm<<<gpro, 128, 0, stream>>>(attnout, projw_t, proj_b, NTOK, CDIM, CDIM,
                                   EPI_PROJ, 0, nullptr, out, x1);
  k_gemm<<<gpro, 128, 0, stream>>>(attnout + (size_t)NTOK * CDIM, projw_t, proj_b,
                                   NTOK, CDIM, CDIM, EPI_PROJ, 1, nullptr,
                                   out + (size_t)NTOK * CDIM, x2);

  // 6. LN2 (identity row order)
  k_ln<<<NTOK, 256, 0, stream>>>(out, n2g, n2b, wbuf, 0);
  k_ln<<<NTOK, 256, 0, stream>>>(out + (size_t)NTOK * CDIM, n2g, n2b,
                                 wbuf + (size_t)NTOK * CDIM, 0);

  // 7. fc1 + GELU -> fp16 hidden (aliases dead QKV/attn region)
  dim3 gfc1(NTOK / 64, HIDN / 64);
  k_gemm<<<gfc1, 128, 0, stream>>>(wbuf, fc1w_t, fc1_b, NTOK, CDIM, HIDN,
                                   EPI_FC1, 0, hidden, nullptr, nullptr);
  k_gemm<<<gfc1, 128, 0, stream>>>(wbuf + (size_t)NTOK * CDIM, fc1w_t, fc1_b,
                                   NTOK, CDIM, HIDN, EPI_FC1, 1,
                                   hidden + (size_t)NTOK * HIDN, nullptr, nullptr);

  // 8. fc2 accumulated into d_out (residual)
  k_gemm<<<gpro, 128, 0, stream>>>(hidden, fc2w_t, fc2_b, NTOK, HIDN, CDIM,
                                   EPI_FC2, 0, nullptr, out, nullptr);
  k_gemm<<<gpro, 128, 0, stream>>>(hidden + (size_t)NTOK * HIDN, fc2w_t, fc2_b,
                                   NTOK, HIDN, CDIM, EPI_FC2, 1, nullptr,
                                   out + (size_t)NTOK * CDIM, nullptr);
}
